// GNNPropertyPredictor_43774306680929
// MI455X (gfx1250) — compile-verified, matrix operand order checked by reading, not run on hardware
//
#include <hip/hip_runtime.h>
#include <hip/hip_bf16.h>
#include <hip/hip_fp16.h>

typedef __attribute__((ext_vector_type(16))) _Float16 v16h;
typedef __attribute__((ext_vector_type(8)))  float    v8f;

#define N_NODES   2048
#define N_EDGES   4096
#define N_GRAPHS  32
#define NPG       64      // nodes per graph (contiguous batch layout)
#define ATOM_DIM  79
#define BOND_DIM  12
#define D         128
#define N_LAYERS  6
#define N_TARGETS 6
#define S2S_ITERS 6

#define KSTEPS 516        // 512 weight K-steps (K=16384) + 4 bias K-steps (K=128), chunks of 32
#define NTILES 8          // 8 x 16 = 128 output dims
#define PACKED_PER_LAYER ((size_t)KSTEPS * NTILES * 32 * 16)   // halves per layer

#define G_KSTEPS 8        // GRU gate GEMM: K = 256 ([agg|h]) in chunks of 32
#define G_NTILES 32       // 512 gate columns: [ir+hr | iz+hz | inn | hn]
#define GPACK_PER_LAYER ((size_t)G_KSTEPS * G_NTILES * 32 * 16)  // halves per layer

__device__ __forceinline__ float gelu_exact(float x) {
    return 0.5f * x * (1.0f + erff(x * 0.70710678118654752f));
}
__device__ __forceinline__ float sigmoidf_(float x) { return 1.0f / (1.0f + expf(-x)); }

// ---------------------------------------------------------------------------
// Pack eW2 (+ eb2 as extra K rows) into WMMA-B fragment layout, f32 -> f16.
// B fragment for v_wmma_f32_16x16x32_f16 (wave32): lane = hi*16+n, elem=2r+o
// holds B[K0 + kb + o, n] with kb = (r>=4?16:0) + 2*(r&3) + hi*8.
// Flat K order: kj = k*128 + j, W2'[(k,j), i] = eW2[k, i*128 + j].
// ---------------------------------------------------------------------------
__global__ void pack_w2_kernel(const float* __restrict__ eW2, const float* __restrict__ eb2,
                               _Float16* __restrict__ w2t) {
    size_t idx = (size_t)blockIdx.x * blockDim.x + threadIdx.x;
    size_t total = (size_t)N_LAYERS * PACKED_PER_LAYER;
    if (idx >= total) return;
    int    l    = (int)(idx / PACKED_PER_LAYER);
    size_t r0   = idx % PACKED_PER_LAYER;
    int    step = (int)(r0 / (NTILES * 32 * 16));
    int    r1   = (int)(r0 % (NTILES * 32 * 16));
    int    nt   = r1 >> 9;
    int    lane = (r1 >> 4) & 31;
    int    elem = r1 & 15;
    int hi = lane >> 4, n = lane & 15;
    int rr = elem >> 1, o = elem & 1;
    int kb = ((rr >= 4) ? 16 : 0) + 2 * (rr & 3) + hi * 8;
    int i  = nt * 16 + n;
    float val;
    if (step < 512) {
        int kj = step * 32 + kb + o;
        int k = kj >> 7, j = kj & 127;
        val = eW2[(size_t)l * (D * D * D) + (size_t)k * (D * D) + (size_t)i * D + j];
    } else {
        int jb = (step - 512) * 32 + kb + o;
        val = eb2[(size_t)l * (D * D) + (size_t)i * D + jb];
    }
    w2t[idx] = (_Float16)val;
}

// ---------------------------------------------------------------------------
// Pack GRU gate weights into WMMA-B layout. Output cols (512):
//   [0,128):   ir+hr  -> Wih row c    (K<128) | Whh row c     (K>=128)
//   [128,256): iz+hz  -> Wih row c    (K<128) | Whh row c     (K>=128)
//   [256,384): inn    -> Wih row c    (K<128) | 0             (K>=128)
//   [384,512): hn     -> 0            (K<128) | Whh row c-128 (K>=128)
// ---------------------------------------------------------------------------
__global__ void pack_wg_kernel(const float* __restrict__ Wih, const float* __restrict__ Whh,
                               _Float16* __restrict__ wg) {
    size_t idx = (size_t)blockIdx.x * blockDim.x + threadIdx.x;
    size_t total = (size_t)N_LAYERS * GPACK_PER_LAYER;
    if (idx >= total) return;
    int    l    = (int)(idx / GPACK_PER_LAYER);
    size_t r0   = idx % GPACK_PER_LAYER;
    int    step = (int)(r0 / (G_NTILES * 32 * 16));
    int    r1   = (int)(r0 % (G_NTILES * 32 * 16));
    int    nt   = r1 >> 9;
    int    lane = (r1 >> 4) & 31;
    int    elem = r1 & 15;
    int hi = lane >> 4, n = lane & 15;
    int rr = elem >> 1, o = elem & 1;
    int kb = ((rr >= 4) ? 16 : 0) + 2 * (rr & 3) + hi * 8;
    int col = nt * 16 + n;
    int kj  = step * 32 + kb + o;          // 0..255
    const size_t lb = (size_t)l * 3 * D * D;
    float val = 0.0f;
    if (col < 256) {
        val = (kj < 128) ? Wih[lb + (size_t)col * D + kj]
                         : Whh[lb + (size_t)col * D + (kj - 128)];
    } else if (col < 384) {
        if (kj < 128) val = Wih[lb + (size_t)col * D + kj];
    } else {
        if (kj >= 128) val = Whh[lb + (size_t)(col - 128) * D + (kj - 128)];
    }
    wg[idx] = (_Float16)val;
}

// ---------------------------------------------------------------------------
// Atom embedding: h = GELU(LayerNorm(x @ W + b)); writes f32 + f16 copies.
// ---------------------------------------------------------------------------
__global__ void embed_kernel(const float* __restrict__ x, const float* __restrict__ W,
                             const float* __restrict__ b, const float* __restrict__ g,
                             const float* __restrict__ be, float* __restrict__ H,
                             _Float16* __restrict__ H16) {
    __shared__ float xs[ATOM_DIM];
    __shared__ float buf[D];
    int n = blockIdx.x, d = threadIdx.x;
    if (d < ATOM_DIM) xs[d] = x[n * ATOM_DIM + d];
    __syncthreads();
    float acc = b[d];
    for (int k = 0; k < ATOM_DIM; ++k) acc += xs[k] * W[k * D + d];
    buf[d] = acc; __syncthreads();
    for (int s = 64; s > 0; s >>= 1) { if (d < s) buf[d] += buf[d + s]; __syncthreads(); }
    float mu = buf[0] * (1.0f / D); __syncthreads();
    float c = acc - mu;
    buf[d] = c * c; __syncthreads();
    for (int s = 64; s > 0; s >>= 1) { if (d < s) buf[d] += buf[d + s]; __syncthreads(); }
    float var = buf[0] * (1.0f / D); __syncthreads();
    float y = c * rsqrtf(var + 1e-5f) * g[d] + be[d];
    float hv = gelu_exact(y);
    H[n * D + d] = hv;
    H16[n * D + d] = (_Float16)hv;
}

// ---------------------------------------------------------------------------
// t = GELU(edge_attr @ eW1[l] + eb1[l]), stored f16 row-major (E x 128)
// ---------------------------------------------------------------------------
__global__ void edge_feat_kernel(const float* __restrict__ ea, const float* __restrict__ W1,
                                 const float* __restrict__ b1, _Float16* __restrict__ T16, int l) {
    __shared__ float es[BOND_DIM];
    int e = blockIdx.x, d = threadIdx.x;
    if (d < BOND_DIM) es[d] = ea[e * BOND_DIM + d];
    __syncthreads();
    const float* Wl = W1 + (size_t)l * BOND_DIM * D;
    float acc = b1[l * D + d];
    for (int k = 0; k < BOND_DIM; ++k) acc += es[k] * Wl[k * D + d];
    T16[(size_t)e * D + d] = (_Float16)gelu_exact(acc);
}

__global__ void zero_kernel(float* __restrict__ p, int n) {
    int i = blockIdx.x * blockDim.x + threadIdx.x;
    if (i < n) p[i] = 0.0f;
}

// ---------------------------------------------------------------------------
// Core WMMA kernel. 4 waves per block; each wave32 owns one 16-edge tile and
// the full 128 output dims:
//   msg[e,i] = sum_{kj} (t[e,k] * h_src[e,j]) * W2'[(k,j), i]   (+ eb2 rows)
// A fragment built on the fly: splat(t) * preloaded h_src fragment.
// B fragments stream with a ping-pong register pipeline (groups of 4).
// Scatter: global_atomic_add_f32 into agg[dst[e], :].
// ---------------------------------------------------------------------------
__global__ void __launch_bounds__(128)
edge_msg_kernel(const _Float16* __restrict__ H16, const _Float16* __restrict__ T16,
                const _Float16* __restrict__ w2L,
                const long long* __restrict__ srcIdx, const long long* __restrict__ dstIdx,
                float* __restrict__ AGG) {
    __shared__ __align__(16) _Float16 sT[4][16 * D];
    __shared__ __align__(16) _Float16 sH[4][16 * D];
    const int lane = threadIdx.x & 31;
    const int w    = threadIdx.x >> 5;
    const int m = lane & 15, hi = lane >> 4;
    const int base = (blockIdx.x * 4 + w) * 16;

    const uint4* tsrc = (const uint4*)(T16 + (size_t)base * D);
    uint4* tdst = (uint4*)sT[w];
    uint4* hdst = (uint4*)sH[w];
    for (int q = 0; q < 8; ++q) {
        int i = q * 32 + lane;
        tdst[i] = tsrc[i];
        int row = i >> 4, col = i & 15;
        long long sr = srcIdx[base + row];
        hdst[i] = ((const uint4*)(H16 + (size_t)sr * D))[col];
    }
    __syncthreads();

    union FragA { v16h v; unsigned u[8]; };
    FragA hf[4];
    const unsigned* sHu = (const unsigned*)sH[w];
    #pragma unroll
    for (int jc = 0; jc < 4; ++jc) {
        #pragma unroll
        for (int rr = 0; rr < 8; ++rr) {
            int kb = ((rr >= 4) ? 16 : 0) + 2 * (rr & 3) + hi * 8;
            int j = jc * 32 + kb;
            hf[jc].u[rr] = sHu[(m * D + j) >> 1];
        }
    }

    v8f acc[NTILES] = {};
    const _Float16* wbase = w2L + (size_t)lane * 16;

    #define LDGRP(dst, g)                                                      \
        do {                                                                   \
            const _Float16* wp_ = wbase + (size_t)(g) * 2048;                  \
            dst[0] = *(const v16h*)(wp_);                                      \
            dst[1] = *(const v16h*)(wp_ + 512);                                \
            dst[2] = *(const v16h*)(wp_ + 1024);                               \
            dst[3] = *(const v16h*)(wp_ + 1536);                               \
        } while (0)

    v16h bA[4], bB[4];
    LDGRP(bA, 0);

    const _Float16* sTw = sT[w];
    for (int kk = 0; kk < 129; ++kk) {
        int ki = kk & 127;
        _Float16 tl = sTw[m * D + ki];
        _Float16 tv = (kk < 128) ? tl : (_Float16)1.0f;
        #pragma unroll
        for (int jc = 0; jc < 4; ++jc) {
            v16h a = hf[jc].v * tv;
            int g = (kk * 4 + jc) * 2;
            __builtin_prefetch(wbase + (size_t)(g + 16) * 2048, 0, 1);
            LDGRP(bB, g + 1);
            acc[0] = __builtin_amdgcn_wmma_f32_16x16x32_f16(false, a, false, bA[0],
                                                            (short)0, acc[0], false, false);
            acc[1] = __builtin_amdgcn_wmma_f32_16x16x32_f16(false, a, false, bA[1],
                                                            (short)0, acc[1], false, false);
            acc[2] = __builtin_amdgcn_wmma_f32_16x16x32_f16(false, a, false, bA[2],
                                                            (short)0, acc[2], false, false);
            acc[3] = __builtin_amdgcn_wmma_f32_16x16x32_f16(false, a, false, bA[3],
                                                            (short)0, acc[3], false, false);
            LDGRP(bA, g + 2);   // final overrun: harmless dead load inside ws
            acc[4] = __builtin_amdgcn_wmma_f32_16x16x32_f16(false, a, false, bB[0],
                                                            (short)0, acc[4], false, false);
            acc[5] = __builtin_amdgcn_wmma_f32_16x16x32_f16(false, a, false, bB[1],
                                                            (short)0, acc[5], false, false);
            acc[6] = __builtin_amdgcn_wmma_f32_16x16x32_f16(false, a, false, bB[2],
                                                            (short)0, acc[6], false, false);
            acc[7] = __builtin_amdgcn_wmma_f32_16x16x32_f16(false, a, false, bB[3],
                                                            (short)0, acc[7], false, false);
        }
    }
    #undef LDGRP

    const int n = lane & 15;
    #pragma unroll
    for (int rr = 0; rr < 8; ++rr) {
        int erow = rr + 8 * hi;
        long long dn = dstIdx[base + erow];
        float* arow = AGG + (size_t)dn * D;
        #pragma unroll
        for (int nt = 0; nt < NTILES; ++nt)
            unsafeAtomicAdd(&arow[nt * 16 + n], acc[nt][rr]);
    }
}

// ---------------------------------------------------------------------------
// GRU gate GEMM via WMMA: Gates(2048 x 512) = [agg | h](f16, 2048 x 256) @ WG.
// One 16-node tile per block; 4 waves x 8 N-tiles cover all 512 gate columns.
// ---------------------------------------------------------------------------
__global__ void __launch_bounds__(128)
gru_gates_kernel(const float* __restrict__ AGG, const _Float16* __restrict__ H16,
                 const _Float16* __restrict__ wgL, float* __restrict__ GATES) {
    __shared__ __align__(16) _Float16 sX[16 * 256];   // [agg | h] tile, 8 KB
    const int tid = threadIdx.x;
    const int lane = tid & 31, w = tid >> 5;
    const int m = lane & 15, hi = lane >> 4;
    const int nodeBase = blockIdx.x * 16;

    for (int q = 0; q < 16; ++q) {
        int idx = q * 128 + tid;               // 0..2047
        int row = idx >> 7, col = idx & 127;
        sX[row * 256 + col]       = (_Float16)AGG[(size_t)(nodeBase + row) * D + col];
        sX[row * 256 + 128 + col] = H16[(size_t)(nodeBase + row) * D + col];
    }
    __syncthreads();

    union FragA { v16h v; unsigned u[8]; };
    FragA af[G_KSTEPS];
    const unsigned* sXu = (const unsigned*)sX;
    #pragma unroll
    for (int s = 0; s < G_KSTEPS; ++s) {
        #pragma unroll
        for (int rr = 0; rr < 8; ++rr) {
            int kb = ((rr >= 4) ? 16 : 0) + 2 * (rr & 3) + hi * 8;
            int k = s * 32 + kb;
            af[s].u[rr] = sXu[(m * 256 + k) >> 1];
        }
    }

    v8f acc[8] = {};
    const _Float16* wb = wgL + (size_t)lane * 16;
    #pragma unroll
    for (int s = 0; s < G_KSTEPS; ++s) {
        #pragma unroll
        for (int q = 0; q < 8; ++q) {
            int nt = w * 8 + q;
            v16h bfr = *(const v16h*)(wb + ((size_t)s * G_NTILES + nt) * 512);
            acc[q] = __builtin_amdgcn_wmma_f32_16x16x32_f16(false, af[s].v, false, bfr,
                                                            (short)0, acc[q], false, false);
        }
    }

    const int n = lane & 15;
    #pragma unroll
    for (int rr = 0; rr < 8; ++rr) {
        int node = nodeBase + rr + 8 * hi;
        float* grow = GATES + (size_t)node * 512;
        #pragma unroll
        for (int q = 0; q < 8; ++q)
            grow[(w * 8 + q) * 16 + n] = acc[q][rr];
    }
}

// ---------------------------------------------------------------------------
// Elementwise GRU blend + LayerNorm + residual from precomputed gates.
// ---------------------------------------------------------------------------
__global__ void gru_elem_kernel(const float* __restrict__ GATES, float* __restrict__ H,
                                _Float16* __restrict__ H16,
                                const float* __restrict__ bih, const float* __restrict__ bhh,
                                const float* __restrict__ lng, const float* __restrict__ lnb,
                                int l) {
    __shared__ float buf[D];
    int node = blockIdx.x, d = threadIdx.x;
    const float* bi = bih + l * 3 * D;
    const float* bh = bhh + l * 3 * D;
    const float* gr = GATES + (size_t)node * 512;
    float hold = H[node * D + d];
    float rz  = gr[d]       + bi[d]       + bh[d];
    float zz  = gr[128 + d] + bi[D + d]   + bh[D + d];
    float inn = gr[256 + d] + bi[2 * D + d];
    float hn  = gr[384 + d] + bh[2 * D + d];
    float r = sigmoidf_(rz), z = sigmoidf_(zz);
    float nn = tanhf(inn + r * hn);
    float hnew = (1.0f - z) * nn + z * hold;
    buf[d] = hnew; __syncthreads();
    for (int s = 64; s > 0; s >>= 1) { if (d < s) buf[d] += buf[d + s]; __syncthreads(); }
    float mu = buf[0] * (1.0f / D); __syncthreads();
    float c = hnew - mu;
    buf[d] = c * c; __syncthreads();
    for (int s = 64; s > 0; s >>= 1) { if (d < s) buf[d] += buf[d + s]; __syncthreads(); }
    float var = buf[0] * (1.0f / D); __syncthreads();
    float out = c * rsqrtf(var + 1e-5f) * lng[l * D + d] + lnb[l * D + d] + hold;
    H[node * D + d] = out;
    H16[node * D + d] = (_Float16)out;
}

// --------------------------- Set2Set readout -------------------------------
__global__ void s2s_scores_kernel(const float* __restrict__ H, const float* __restrict__ HG,
                                  const long long* __restrict__ batch, float* __restrict__ SC) {
    int nidx = blockIdx.x * blockDim.x + threadIdx.x;
    if (nidx >= N_NODES) return;
    int g = (int)batch[nidx];
    float s = 0.f;
    for (int d = 0; d < D; ++d) s += H[nidx * D + d] * HG[g * D + d];
    SC[nidx] = s;
}

__global__ void s2s_softmax_kernel(const float* __restrict__ SC, float* __restrict__ WGT) {
    __shared__ float buf[NPG];
    int g = blockIdx.x, t = threadIdx.x;
    int nidx = g * NPG + t;
    float s = SC[nidx];
    buf[t] = s; __syncthreads();
    for (int k = 32; k > 0; k >>= 1) { if (t < k) buf[t] = fmaxf(buf[t], buf[t + k]); __syncthreads(); }
    float mx = fmaxf(buf[0], 0.0f); __syncthreads();
    float e = expf(s - mx);
    buf[t] = e; __syncthreads();
    for (int k = 32; k > 0; k >>= 1) { if (t < k) buf[t] += buf[t + k]; __syncthreads(); }
    float ssum = buf[0];
    WGT[nidx] = e / (ssum + 1e-8f);
}

__global__ void s2s_readout_kernel(const float* __restrict__ H, const float* __restrict__ WGT,
                                   float* __restrict__ RO, float* __restrict__ QS) {
    int g = blockIdx.x, d = threadIdx.x;
    float acc = 0.f;
    for (int q = 0; q < NPG; ++q) acc += WGT[g * NPG + q] * H[(size_t)(g * NPG + q) * D + d];
    RO[g * D + d] = acc;
    QS[g * 2 * D + D + d] = acc;   // second half of q_star = readout
}

__global__ void s2s_lstm_kernel(const float* __restrict__ RO, float* __restrict__ HG,
                                float* __restrict__ CG, float* __restrict__ QS,
                                const float* __restrict__ Wih, const float* __restrict__ Whh,
                                const float* __restrict__ bih, const float* __restrict__ bhh) {
    __shared__ float hgs[D], ros[D];
    int g = blockIdx.x, d = threadIdx.x;
    hgs[d] = HG[g * D + d];
    ros[d] = RO[g * D + d];
    __syncthreads();
    float gi = bih[d] + bhh[d];
    float gf = bih[D + d] + bhh[D + d];
    float gg = bih[2 * D + d] + bhh[2 * D + d];
    float go = bih[3 * D + d] + bhh[3 * D + d];
    for (int k = 0; k < D; ++k) {
        float rv = ros[k], hv = hgs[k];
        gi += Wih[d * D + k] * rv + Whh[d * D + k] * hv;
        gf += Wih[(D + d) * D + k] * rv + Whh[(D + d) * D + k] * hv;
        gg += Wih[(2 * D + d) * D + k] * rv + Whh[(2 * D + d) * D + k] * hv;
        go += Wih[(3 * D + d) * D + k] * rv + Whh[(3 * D + d) * D + k] * hv;
    }
    float cn = sigmoidf_(gf) * CG[g * D + d] + sigmoidf_(gi) * tanhf(gg);
    float hnew = sigmoidf_(go) * tanhf(cn);
    HG[g * D + d] = hnew;
    CG[g * D + d] = cn;
    QS[g * 2 * D + d] = hnew;      // first half of q_star = hg
}

// --------------------------- Final MLP + heads -----------------------------
__global__ void final_kernel(const float* __restrict__ QS, const float* __restrict__ Ws,
                             const float* __restrict__ bs, const float* __restrict__ Wh,
                             const float* __restrict__ bh, float* __restrict__ out) {
    __shared__ float qs[2 * D];
    __shared__ float sh[D];
    int g = blockIdx.x, d = threadIdx.x;
    qs[d] = QS[g * 2 * D + d];
    qs[D + d] = QS[g * 2 * D + D + d];
    __syncthreads();
    float acc = bs[d];
    for (int k = 0; k < 2 * D; ++k) acc += qs[k] * Ws[k * D + d];
    sh[d] = gelu_exact(acc);
    __syncthreads();
    if (d < N_TARGETS) {
        float o = bh[d];
        for (int k = 0; k < D; ++k) o += sh[k] * Wh[k * N_TARGETS + d];
        out[g * N_TARGETS + d] = o;
    }
}

// ---------------------------------------------------------------------------
extern "C" void kernel_launch(void* const* d_in, const int* in_sizes, int n_in,
                              void* d_out, int out_size, void* d_ws, size_t ws_size,
                              hipStream_t stream) {
    const float*     x       = (const float*)d_in[0];
    const long long* ei      = (const long long*)d_in[1];
    const float*     ea      = (const float*)d_in[2];
    const long long* batch   = (const long long*)d_in[3];
    const float*     W_embed = (const float*)d_in[4];
    const float*     b_embed = (const float*)d_in[5];
    const float*     lng_e   = (const float*)d_in[6];
    const float*     lnb_e   = (const float*)d_in[7];
    const float*     eW1     = (const float*)d_in[8];
    const float*     eb1     = (const float*)d_in[9];
    const float*     eW2     = (const float*)d_in[10];
    const float*     eb2     = (const float*)d_in[11];
    const float*     gWih    = (const float*)d_in[12];
    const float*     gWhh    = (const float*)d_in[13];
    const float*     gbih    = (const float*)d_in[14];
    const float*     gbhh    = (const float*)d_in[15];
    const float*     ln_g    = (const float*)d_in[16];
    const float*     ln_b    = (const float*)d_in[17];
    const float*     lWih    = (const float*)d_in[18];
    const float*     lWhh    = (const float*)d_in[19];
    const float*     lbih    = (const float*)d_in[20];
    const float*     lbhh    = (const float*)d_in[21];
    const float*     W_sh    = (const float*)d_in[22];
    const float*     b_sh    = (const float*)d_in[23];
    const float*     W_hd    = (const float*)d_in[24];
    const float*     b_hd    = (const float*)d_in[25];
    float* outp = (float*)d_out;

    // Workspace carve-up (~35 MB)
    char* ws = (char*)d_ws;
    size_t off = 0;
    _Float16* W2T = (_Float16*)(ws + off); off += N_LAYERS * PACKED_PER_LAYER * sizeof(_Float16);
    _Float16* WG  = (_Float16*)(ws + off); off += N_LAYERS * GPACK_PER_LAYER * sizeof(_Float16);
    _Float16* T16 = (_Float16*)(ws + off); off += (size_t)N_EDGES * D * sizeof(_Float16);
    _Float16* H16 = (_Float16*)(ws + off); off += (size_t)N_NODES * D * sizeof(_Float16);
    float* H     = (float*)(ws + off); off += (size_t)N_NODES * D * sizeof(float);
    float* AGG   = (float*)(ws + off); off += (size_t)N_NODES * D * sizeof(float);
    float* GATES = (float*)(ws + off); off += (size_t)N_NODES * 512 * sizeof(float);
    float* SC  = (float*)(ws + off); off += (size_t)N_NODES * sizeof(float);
    float* WGT = (float*)(ws + off); off += (size_t)N_NODES * sizeof(float);
    float* HG  = (float*)(ws + off); off += (size_t)N_GRAPHS * D * sizeof(float);
    float* CG  = (float*)(ws + off); off += (size_t)N_GRAPHS * D * sizeof(float);
    float* RO  = (float*)(ws + off); off += (size_t)N_GRAPHS * D * sizeof(float);
    float* QS  = (float*)(ws + off); off += (size_t)N_GRAPHS * 2 * D * sizeof(float);

    const long long* srcIdx = ei;
    const long long* dstIdx = ei + N_EDGES;

    {   // repack edge-network weights into WMMA-B fragment layout (f16)
        size_t total = (size_t)N_LAYERS * PACKED_PER_LAYER;
        pack_w2_kernel<<<(int)((total + 255) / 256), 256, 0, stream>>>(eW2, eb2, W2T);
    }
    {   // repack GRU gate weights
        size_t total = (size_t)N_LAYERS * GPACK_PER_LAYER;
        pack_wg_kernel<<<(int)((total + 255) / 256), 256, 0, stream>>>(gWih, gWhh, WG);
    }

    embed_kernel<<<N_NODES, D, 0, stream>>>(x, W_embed, b_embed, lng_e, lnb_e, H, H16);

    for (int l = 0; l < N_LAYERS; ++l) {
        edge_feat_kernel<<<N_EDGES, D, 0, stream>>>(ea, eW1, eb1, T16, l);
        zero_kernel<<<(N_NODES * D + 255) / 256, 256, 0, stream>>>(AGG, N_NODES * D);
        edge_msg_kernel<<<N_EDGES / 64, 128, 0, stream>>>(
            H16, T16, W2T + (size_t)l * PACKED_PER_LAYER, srcIdx, dstIdx, AGG);
        gru_gates_kernel<<<N_NODES / 16, 128, 0, stream>>>(
            AGG, H16, WG + (size_t)l * GPACK_PER_LAYER, GATES);
        gru_elem_kernel<<<N_NODES, D, 0, stream>>>(GATES, H, H16, gbih, gbhh,
                                                   ln_g, ln_b, l);
    }

    zero_kernel<<<(N_GRAPHS * D + 255) / 256, 256, 0, stream>>>(HG, N_GRAPHS * D);
    zero_kernel<<<(N_GRAPHS * D + 255) / 256, 256, 0, stream>>>(CG, N_GRAPHS * D);
    for (int it = 0; it < S2S_ITERS; ++it) {
        s2s_scores_kernel<<<(N_NODES + 255) / 256, 256, 0, stream>>>(H, HG, batch, SC);
        s2s_softmax_kernel<<<N_GRAPHS, NPG, 0, stream>>>(SC, WGT);
        s2s_readout_kernel<<<N_GRAPHS, D, 0, stream>>>(H, WGT, RO, QS);
        s2s_lstm_kernel<<<N_GRAPHS, D, 0, stream>>>(RO, HG, CG, QS, lWih, lWhh, lbih, lbhh);
    }

    final_kernel<<<N_GRAPHS, D, 0, stream>>>(QS, W_sh, b_sh, W_hd, b_hd, outp);
}
